// Stride_Self_Attn_67869073211862
// MI455X (gfx1250) — matrix operand order repaired and checked by reading, not compiled, not run on hardware
//
#include <hip/hip_runtime.h>

// ---------------------------------------------------------------------------
// CDNA5 (gfx1250) wave32 WMMA implementation of the 3-branch strided self-attn.
// All GEMMs via v_wmma_f32_16x16x32_f16. Workspace layout keeps the full
// energy/att buffer (134MB) + frags inside the 192MB L2.
// ---------------------------------------------------------------------------

typedef __attribute__((ext_vector_type(16))) _Float16 v16h;
typedef __attribute__((ext_vector_type(8)))  float    v8f;

union F16x16 { v16h v; _Float16 h[16]; };
union F32x8  { v8f  v; float    f[8];  };
union HPair  { unsigned int u; _Float16 h[2]; };

#define BATCH 4
#define CHAN  256
#define CH8   32

__device__ __forceinline__ v8f wmma_f16(v16h a, v16h b, v8f c) {
  return __builtin_amdgcn_wmma_f32_16x16x32_f16(false, a, false, b, (short)0, c,
                                                false, false);
}

// ---------------------------------------------------------------------------
// MaxPool2d over (W,H) dims of x:(B,C,64,64), -inf padding (torch semantics).
// ---------------------------------------------------------------------------
__global__ void maxpool_k(const float* __restrict__ x, float* __restrict__ y,
                          int Wo, int Ho, int kw, int kh, int sw, int sh,
                          int pw, int ph) {
  int idx = blockIdx.x * blockDim.x + threadIdx.x;
  int total = BATCH * CHAN * Wo * Ho;
  if (idx >= total) return;
  int h = idx % Ho; int t = idx / Ho;
  int w = t % Wo;   t /= Wo;
  int c = t % CHAN; int b = t / CHAN;
  const float* xp = x + ((size_t)(b * CHAN + c)) * 64 * 64;
  float m = -__builtin_inff();
  int w0 = w * sw - pw, h0 = h * sh - ph;
  for (int i = 0; i < kw; ++i) {
    int ww = w0 + i; if (ww < 0 || ww >= 64) continue;
    for (int j = 0; j < kh; ++j) {
      int hh = h0 + j; if (hh < 0 || hh >= 64) continue;
      m = fmaxf(m, xp[ww * 64 + hh]);
    }
  }
  y[idx] = m;
}

// ---------------------------------------------------------------------------
// conv1x1 projections as WMMA GEMMs: D(16o x 16n) = sum_{8 chunks} W(16o x 32c)
// * X(32c x 16n).  Results scattered into fragment-ready layouts:
//   qA: A-frag  (M=n rows, K=o)   [b][ntile][lane32][16]
//   kB: B-frag  (K=o, N=m)        [b][mtile][lane32][16]
//   vB: B-frag  (K=m, N=c)        [b][mchunk][ctile][lane32][16]
// 20 sub-jobs per (b, ntile): 2 q-tiles, 2 k-tiles, 16 v-tiles.
// ---------------------------------------------------------------------------
__global__ void proj_k(const float* __restrict__ xsrc, int N,
                       const float* __restrict__ wq, const float* __restrict__ bq,
                       const float* __restrict__ wk, const float* __restrict__ bk,
                       const float* __restrict__ wv, const float* __restrict__ bv,
                       _Float16* __restrict__ qA, _Float16* __restrict__ kB,
                       _Float16* __restrict__ vB) {
  int gw   = blockIdx.x * (blockDim.x >> 5) + (threadIdx.x >> 5);
  int lane = threadIdx.x & 31;
  int ntiles = N >> 4;
  int sub = gw % 20; int t = gw / 20;
  int ntile = t % ntiles; int b = t / ntiles;
  if (b >= BATCH) return;

  const float* wsel; const float* bsel; int otile;
  if (sub < 2)      { wsel = wq; bsel = bq; otile = sub;     }
  else if (sub < 4) { wsel = wk; bsel = bk; otile = sub - 2; }
  else              { wsel = wv; bsel = bv; otile = sub - 4; }

  int nloc = lane & 15, hi = lane >> 4;
  int ncol = ntile * 16 + nloc;
  int orow = otile * 16 + nloc;
  const float* xb = xsrc + (size_t)b * CHAN * N;

  F32x8 acc = {};
  for (int chunk = 0; chunk < CHAN / 32; ++chunk) {
    F16x16 a, bb;
#pragma unroll
    for (int e = 0; e < 16; ++e) {
      // A-frag element (row=o, K): K = (e&7) + 8*hi + 16*(e>>3)
      int Ka = (e & 7) + 8 * hi + 16 * (e >> 3);
      a.h[e] = (_Float16)wsel[orow * CHAN + chunk * 32 + Ka];
      // B-frag element (K, col=n): K = (e&7) + 8*hi + 16*(e>>3)
      int Kb = (e & 7) + 8 * hi + 16 * (e >> 3);
      bb.h[e] = (_Float16)xb[(size_t)(chunk * 32 + Kb) * N + ncol];
    }
    acc.v = wmma_f16(a.v, bb.v, acc.v);
  }

#pragma unroll
  for (int r = 0; r < 8; ++r) {
    int o = otile * 16 + r + 8 * hi;   // D row
    int n = ncol;                      // D col
    _Float16 hv = (_Float16)(acc.f[r] + bsel[o]);
    if (sub < 2) {            // q -> A-frag, element (M=n%16, K=o)
      int l2 = nloc + 16 * ((o >> 3) & 1);
      int h2 = (o & 7) + 8 * (o >> 4);
      qA[(((size_t)b * ntiles + ntile) * 32 + l2) * 16 + h2] = hv;
    } else if (sub < 4) {     // k -> B-frag, element (K=o, N=n%16)
      int l2 = nloc + 16 * ((o >> 3) & 1);
      int h2 = (o & 7) + 8 * (o >> 4);
      kB[(((size_t)b * ntiles + ntile) * 32 + l2) * 16 + h2] = hv;
    } else {                  // v^T -> B-frag, element (K=n%32, N=o%16)
      int chnk = n >> 5, km = n & 31, ctile = o >> 4, cl = o & 15;
      int l2 = cl + 16 * ((km >> 3) & 1);
      int h2 = (km & 7) + 8 * (km >> 4);
      vB[((((size_t)b * (N >> 5) + chnk) * 16 + ctile) * 32 + l2) * 16 + h2] = hv;
    }
  }
}

// ---------------------------------------------------------------------------
// energy[b,n,m] = sum_c8 q[c8,n] * k[c8,m].  K=32 exactly -> one WMMA per
// 16x16 tile.  Wave holds its q A-frag (one v16h load) and streams k B-frags.
// Output stored row-major f16.
// ---------------------------------------------------------------------------
__global__ void energy_k(const _Float16* __restrict__ qA,
                         const _Float16* __restrict__ kB,
                         _Float16* __restrict__ energy, int N) {
  int gw   = blockIdx.x * (blockDim.x >> 5) + (threadIdx.x >> 5);
  int lane = threadIdx.x & 31;
  int ntiles = N >> 4;
  int ntile = gw % ntiles, b = gw / ntiles;
  if (b >= BATCH) return;
  int nloc = lane & 15, hi = lane >> 4;

  F16x16 a;
  a.v = *(const v16h*)&qA[(((size_t)b * ntiles + ntile) * 32 + lane) * 16];
  for (int mtile = 0; mtile < ntiles; ++mtile) {
    F16x16 kb;
    kb.v = *(const v16h*)&kB[(((size_t)b * ntiles + mtile) * 32 + lane) * 16];
    F32x8 s = {};
    s.v = wmma_f16(a.v, kb.v, s.v);
#pragma unroll
    for (int r = 0; r < 8; ++r) {
      int n = ntile * 16 + r + 8 * hi;
      int m = mtile * 16 + nloc;
      energy[((size_t)b * N + n) * N + m] = (_Float16)s.f[r];
    }
  }
}

// ---------------------------------------------------------------------------
// Row softmax + in-place relayout into WMMA A-fragments.  One 512-thread block
// per 16-row band (the A-frag band is byte-identical to those 16 energy rows,
// so stage rows in registers, barrier, then overwrite).  Wave32 shfl reduce.
// ---------------------------------------------------------------------------
template <int N>
__global__ void softmax_k(_Float16* __restrict__ eatt) {
  constexpr int NTILES = N >> 4;
  constexpr int NU = N / 64;                 // u32 (=2 halfs) per lane
  int band = blockIdx.x;
  int b = band / NTILES, ntile = band % NTILES;
  int w    = threadIdx.x >> 5;               // row within band (0..15)
  int lane = threadIdx.x & 31;
  int n = ntile * 16 + w;

  const unsigned int* rowp =
      (const unsigned int*)&eatt[((size_t)b * N + n) * N];
  unsigned int rv[NU];
#pragma unroll
  for (int i = 0; i < NU; ++i) rv[i] = rowp[lane + 32 * i];
  __syncthreads();   // whole band captured in registers before any overwrite

  float mx = -__builtin_inff();
#pragma unroll
  for (int i = 0; i < NU; ++i) {
    HPair p; p.u = rv[i];
    mx = fmaxf(mx, fmaxf((float)p.h[0], (float)p.h[1]));
  }
  for (int off = 16; off > 0; off >>= 1) mx = fmaxf(mx, __shfl_xor(mx, off));

  float sum = 0.f;
#pragma unroll
  for (int i = 0; i < NU; ++i) {
    HPair p; p.u = rv[i];
    sum += __expf((float)p.h[0] - mx) + __expf((float)p.h[1] - mx);
  }
  for (int off = 16; off > 0; off >>= 1) sum += __shfl_xor(sum, off);
  float inv = 1.f / sum;

  // A-frag target: element (M=w, K=km).  km0=(2*lane)&31 is even, so the pair
  // lands on one lane at adjacent even/odd halves -> single dword store.
  int km0 = (2 * lane) & 31;
  int l2  = w + 16 * ((km0 >> 3) & 1);
  int h2  = (km0 & 7) + 8 * (km0 >> 4);
  unsigned int* dst =
      (unsigned int*)&eatt[(((size_t)b * NTILES + ntile) * (N >> 5)) * 512];
#pragma unroll
  for (int i = 0; i < NU; ++i) {
    HPair p; p.u = rv[i];
    HPair q2;
    q2.h[0] = (_Float16)(__expf((float)p.h[0] - mx) * inv);
    q2.h[1] = (_Float16)(__expf((float)p.h[1] - mx) * inv);
    int chunk = 2 * i + (lane >> 4);
    dst[(chunk * 512 + l2 * 16 + h2) >> 1] = q2.u;
  }
}

// ---------------------------------------------------------------------------
// outT(16n x 16c) = sum_{m chunks} att(16n x 32m) * v^T(32m x 16c); epilogue
// applies gamma, residual, and nearest-upsample channel-concat store.
// br: 0 = full-res, 1 = height-pooled (repeat W), 2 = width-pooled (repeat H).
// ---------------------------------------------------------------------------
__global__ void attnout_k(const _Float16* __restrict__ attA,
                          const _Float16* __restrict__ vB,
                          const float* __restrict__ xres,
                          const float* __restrict__ gamma,
                          float* __restrict__ out, int N, int br) {
  int gw   = blockIdx.x * (blockDim.x >> 5) + (threadIdx.x >> 5);
  int lane = threadIdx.x & 31;
  int ntiles = N >> 4, nchunks = N >> 5;
  int ctile = gw & 15; int t = gw >> 4;
  int ntile = t % ntiles, b = t / ntiles;
  if (b >= BATCH) return;

  const _Float16* ab = attA + ((size_t)b * ntiles + ntile) * nchunks * 512;
  const _Float16* vb = vB + (size_t)b * nchunks * 16 * 512;
  F32x8 acc = {};
  for (int ch = 0; ch < nchunks; ++ch) {
    v16h a = *(const v16h*)&ab[(size_t)ch * 512 + lane * 16];
    v16h v = *(const v16h*)&vb[((size_t)ch * 16 + ctile) * 512 + lane * 16];
    acc.v = wmma_f16(a, v, acc.v);
  }

  float g = gamma[0];
  int nloc = lane & 15, hi = lane >> 4;
#pragma unroll
  for (int r = 0; r < 8; ++r) {
    int n = ntile * 16 + r + 8 * hi;   // spatial index
    int c = ctile * 16 + nloc;         // channel
    float val = g * acc.f[r] + xres[((size_t)b * CHAN + c) * N + n];
    if (br == 0) {
      int ww = n >> 6, hh = n & 63;
      out[(((size_t)b * 768 + c) * 64 + ww) * 64 + hh] = val;
    } else if (br == 1) {              // (B,C,32,64) -> repeat axis W x2
      int wp = n >> 6, hh = n & 63;
      size_t base = (((size_t)b * 768 + 256 + c) * 64 + 2 * wp) * 64 + hh;
      out[base] = val; out[base + 64] = val;
    } else {                           // (B,C,64,32) -> repeat axis H x2
      int ww = n >> 5, hp = n & 31;
      size_t base = (((size_t)b * 768 + 512 + c) * 64 + ww) * 64 + 2 * hp;
      out[base] = val; out[base + 1] = val;
    }
  }
}

// ---------------------------------------------------------------------------
// Host side
// ---------------------------------------------------------------------------
static void run_branch(const float* xsrc, int N, const float* wq,
                       const float* bq, const float* wk, const float* bk,
                       const float* wv, const float* bv, const float* gamma,
                       _Float16* qA, _Float16* kB, _Float16* vB,
                       _Float16* eatt, float* out, int br, hipStream_t stream) {
  int ntiles = N >> 4;
  int pw = BATCH * ntiles * 20;                 // proj waves
  proj_k<<<pw / 8, 256, 0, stream>>>(xsrc, N, wq, bq, wk, bk, wv, bv,
                                     qA, kB, vB);
  int ew = BATCH * ntiles;                      // energy waves
  energy_k<<<(ew + 7) / 8, 256, 0, stream>>>(qA, kB, eatt, N);
  if (N == 4096) softmax_k<4096><<<BATCH * ntiles, 512, 0, stream>>>(eatt);
  else           softmax_k<2048><<<BATCH * ntiles, 512, 0, stream>>>(eatt);
  int ow = BATCH * ntiles * 16;                 // output waves
  attnout_k<<<ow / 8, 256, 0, stream>>>(eatt, vB, xsrc, gamma, out, N, br);
}

extern "C" void kernel_launch(void* const* d_in, const int* in_sizes, int n_in,
                              void* d_out, int out_size, void* d_ws,
                              size_t ws_size, hipStream_t stream) {
  const float* x   = (const float*)d_in[0];
  const float* wq  = (const float*)d_in[1];  const float* bq  = (const float*)d_in[2];
  const float* wk  = (const float*)d_in[3];  const float* bk  = (const float*)d_in[4];
  const float* wv  = (const float*)d_in[5];  const float* bv  = (const float*)d_in[6];
  const float* wq3 = (const float*)d_in[7];  const float* bq3 = (const float*)d_in[8];
  const float* wk3 = (const float*)d_in[9];  const float* bk3 = (const float*)d_in[10];
  const float* wv3 = (const float*)d_in[11]; const float* bv3 = (const float*)d_in[12];
  const float* wq7 = (const float*)d_in[13]; const float* bq7 = (const float*)d_in[14];
  const float* wk7 = (const float*)d_in[15]; const float* bk7 = (const float*)d_in[16];
  const float* wv7 = (const float*)d_in[17]; const float* bv7 = (const float*)d_in[18];
  const float* gamma  = (const float*)d_in[19];
  const float* gamma3 = (const float*)d_in[20];
  // gamma7 = d_in[21] is intentionally unused: reference passes `gamma` to the
  // width-pooled branch (faithfully replicated).
  float* out = (float*)d_out;

  // Workspace layout (total 154 MiB; energy/att + frags sit inside 192MB L2):
  //   [0,8M)    xh  (B,C,32,64) f32
  //   [8M,16M)  xw  (B,C,64,32) f32
  //   [16M,17M) qA  A-frags f16
  //   [17M,18M) kB  B-frags f16
  //   [18M,26M) vB  B-frags f16
  //   [26M,154M) energy / att (in-place) f16
  char* w0 = (char*)d_ws;
  float*    xh   = (float*)w0;
  float*    xw   = (float*)(w0 + (size_t)(8u << 20));
  _Float16* qA   = (_Float16*)(w0 + (size_t)(16u << 20));
  _Float16* kB   = (_Float16*)(w0 + (size_t)(17u << 20));
  _Float16* vB   = (_Float16*)(w0 + (size_t)(18u << 20));
  _Float16* eatt = (_Float16*)(w0 + (size_t)(26u << 20));

  int ptot = BATCH * CHAN * 32 * 64;   // 2,097,152 outputs per pooled tensor
  maxpool_k<<<ptot / 256, 256, 0, stream>>>(x, xh, 32, 64, 5, 3, 2, 1, 2, 1);
  maxpool_k<<<ptot / 256, 256, 0, stream>>>(x, xw, 64, 32, 3, 5, 1, 2, 1, 2);

  // Branches run sequentially on `stream`, reusing q/k/v/energy regions.
  run_branch(x,  4096, wq,  bq,  wk,  bk,  wv,  bv,  gamma,
             qA, kB, vB, eatt, out, 0, stream);
  run_branch(xh, 2048, wq3, bq3, wk3, bk3, wv3, bv3, gamma3,
             qA, kB, vB, eatt, out, 1, stream);
  run_branch(xw, 2048, wq7, bq7, wk7, bk7, wv7, bv7, gamma,   // NOTE: gamma
             qA, kB, vB, eatt, out, 2, stream);
}
